// EnergyBasedModel_7851200217681
// MI455X (gfx1250) — matrix operand order repaired and checked by reading, not compiled, not run on hardware
//
#include <hip/hip_runtime.h>
#include <hip/hip_bf16.h>
#include <stdint.h>

typedef _Float16 v8h  __attribute__((ext_vector_type(8)));
typedef _Float16 v16h __attribute__((ext_vector_type(16)));
typedef float    v8f  __attribute__((ext_vector_type(8)));

// Model geometry (fixed by the reference)
#define B_ROWS   8192
#define D0       784
#define D0P      896        // 784 padded to a multiple of 128 (staging chunk)
#define H1       1024
#define H2       1024
#define D3       10
#define D3P      16
#define SCOLS    2058       // 1024 + 1024 + 10
#define N_ITER   10
#define DECAY    0.8f       // (1 - DT)
#define DTC      0.2f
#define KCHUNK   128        // K staged per LDS buffer
#define BSH_HALF (128 * KCHUNK)   // halves per LDS buffer (128 N rows x 128 K)

__device__ __forceinline__ float clip01(float v) {
    return fminf(fmaxf(v, 0.0f), 1.0f);
}

// ---------------------------------------------------------------------------
// Prep kernels (run once per launch; cheap)
// ---------------------------------------------------------------------------
__global__ void k_prep_x(const float* __restrict__ X, _Float16* __restrict__ rX) {
    int idx = blockIdx.x * blockDim.x + threadIdx.x;      // B_ROWS * D0P
    int m = idx / D0P, k = idx % D0P;
    if (m >= B_ROWS) return;
    float v = (k < D0) ? clip01(X[(size_t)m * D0 + k]) : 0.0f;
    rX[idx] = (_Float16)v;
}

__global__ void k_prep_w0(const float* __restrict__ W0, _Float16* __restrict__ W0T) {
    int idx = blockIdx.x * blockDim.x + threadIdx.x;      // H1 * D0P
    int n = idx / D0P, k = idx % D0P;
    if (n >= H1) return;
    float v = (k < D0) ? W0[(size_t)k * H1 + n] : 0.0f;   // W0T[n][k] = W0[k][n]
    W0T[idx] = (_Float16)v;
}

__global__ void k_prep_w1(const float* __restrict__ W1,
                          _Float16* __restrict__ W1T,     // W1T[n][k] = W1[k][n]  (B = W1)
                          _Float16* __restrict__ W1R) {   // W1R[n][k] = W1[n][k]  (B = W1^T)
    int idx = blockIdx.x * blockDim.x + threadIdx.x;      // H1 * H2
    if (idx >= H1 * H2) return;
    int n = idx >> 10, k = idx & 1023;
    W1T[idx] = (_Float16)W1[(size_t)k * H2 + n];
    W1R[idx] = (_Float16)W1[idx];
}

// ---------------------------------------------------------------------------
// Per-iteration: snapshot rho = clip(s) into f16 activation buffers
// ---------------------------------------------------------------------------
__global__ void k_actify(const float* __restrict__ S,
                         _Float16* __restrict__ r1,
                         _Float16* __restrict__ r2,
                         _Float16* __restrict__ r3p) {
    int idx = blockIdx.x * blockDim.x + threadIdx.x;      // B_ROWS * 2064
    int m = idx / 2064, c = idx % 2064;
    if (m >= B_ROWS) return;
    const size_t srow = (size_t)m * SCOLS;
    if (c < H1) {
        r1[(size_t)m * H1 + c] = (_Float16)clip01(S[srow + c]);
    } else if (c < 2048) {
        int cc = c - 1024;
        r2[(size_t)m * H2 + cc] = (_Float16)clip01(S[srow + 1024 + cc]);
    } else {
        int cc = c - 2048;
        float v = (cc < D3) ? clip01(S[srow + 2048 + cc]) : 0.0f;
        r3p[(size_t)m * D3P + cc] = (_Float16)v;
    }
}

// ---------------------------------------------------------------------------
// Async B-tile staging: 128 N-rows x 128 K halves (32 KB) per buffer.
// XOR swizzle: 16B slot s of row n stored at slot (s ^ (n & 15)) -> the
// 16-lane fragment reads hit 16 distinct 16B slots = all 64 LDS banks.
// Each thread issues 8 global_load_async_to_lds_b128 (ASYNCcnt += 8/wave).
// ---------------------------------------------------------------------------
__device__ __forceinline__ void issue_stage(const _Float16* __restrict__ BT, int ldb,
                                            int n0, int kc, unsigned ldsBase, int tid) {
#pragma unroll
    for (int i = 0; i < 8; ++i) {
        int linear = i * 256 + tid;          // 0..2047
        int nl = linear >> 4;                // 0..127  (N row)
        int slot = linear & 15;              // 16B slot within row
        const _Float16* g = BT + (size_t)(n0 + nl) * ldb + kc + slot * 8;
        unsigned lo = ldsBase + (unsigned)(nl * 256 + ((slot ^ (nl & 15)) << 4));
        asm volatile("global_load_async_to_lds_b128 %0, %1, off"
                     :: "v"(lo), "v"(g) : "memory");
    }
}

__device__ __forceinline__ void wait_async(int pending) {
    if (pending) asm volatile("s_wait_asynccnt 0x8" ::: "memory");
    else         asm volatile("s_wait_asynccnt 0x0" ::: "memory");
}

// One GEMM pass: acc[2][4] += A(32x K) @ B(K x 64), B staged via async->LDS.
// Block tile 128x128 (8 waves as 4 M x 2 N), wave tile 32x64.
__device__ __forceinline__ void staged_gemm_pass(const _Float16* __restrict__ A, int lda,
                                                 const _Float16* __restrict__ BT, int ldb,
                                                 int K, int mBase, int nWave, int n0,
                                                 int lane, int tid, const _Float16* Bsh,
                                                 unsigned bshBase, v8f acc[2][4]) {
    const bool hi = lane >= 16;
    const _Float16* aRow0 = A + (size_t)(mBase + (lane & 15)) * lda + (hi ? 8 : 0);
    const _Float16* aRow1 = aRow0 + (size_t)16 * lda;
    const int nChunks = K / KCHUNK;

    issue_stage(BT, ldb, n0, 0, bshBase, tid);                    // preload buf 0
    for (int c = 0; c < nChunks; ++c) {
        const int buf = c & 1;
        const int more = (c + 1 < nChunks);
        if (more)
            issue_stage(BT, ldb, n0, (c + 1) * KCHUNK,
                        bshBase + (unsigned)((buf ^ 1) * BSH_HALF * 2), tid);
        wait_async(more);
        __syncthreads();                                          // buf c ready for all waves

        const _Float16* Bbuf = Bsh + buf * BSH_HALF;
        const int kcA = c * KCHUNK;
        __builtin_prefetch(aRow0 + kcA + KCHUNK, 0, 3);
        __builtin_prefetch(aRow1 + kcA + KCHUNK, 0, 3);
#pragma unroll
        for (int ks = 0; ks < 4; ++ks) {
            union { v16h v; v8h h[2]; } a0, a1;
            const _Float16* ap0 = aRow0 + kcA + ks * 32;
            const _Float16* ap1 = aRow1 + kcA + ks * 32;
            a0.h[0] = *(const v8h*)(ap0);      a0.h[1] = *(const v8h*)(ap0 + 16);
            a1.h[0] = *(const v8h*)(ap1);      a1.h[1] = *(const v8h*)(ap1 + 16);
            const int slotBase = ks * 4 + (hi ? 1 : 0);
#pragma unroll
            for (int j = 0; j < 4; ++j) {
                const int nl = nWave + j * 16 + (lane & 15);
                const int sw = nl & 15;
                union { v16h v; v8h h[2]; } b;
                b.h[0] = *(const v8h*)(Bbuf + nl * KCHUNK + (((slotBase    ) ^ sw) << 3));
                b.h[1] = *(const v8h*)(Bbuf + nl * KCHUNK + (((slotBase + 2) ^ sw) << 3));
                acc[0][j] = __builtin_amdgcn_wmma_f32_16x16x32_f16(
                    false, a0.v, false, b.v, (short)0, acc[0][j], false, false);
                acc[1][j] = __builtin_amdgcn_wmma_f32_16x16x32_f16(
                    false, a1.v, false, b.v, (short)0, acc[1][j], false, false);
            }
        }
        __syncthreads();                                          // done reading buf c
    }
}

// Layer-1 pre-activation:  H = 0.8*s1 + 0.2*rho'(s1) * (rX@W0 + r2@W1^T)
__global__ void __launch_bounds__(256)
k_gemm_layer1(const _Float16* __restrict__ rX, const _Float16* __restrict__ W0T,
              const _Float16* __restrict__ r2, const _Float16* __restrict__ W1R,
              const float* __restrict__ S, float* __restrict__ H) {
    __shared__ _Float16 Bsh[2 * BSH_HALF];                        // 64 KB
    const int tid = threadIdx.x, lane = tid & 31, wave = tid >> 5;
    const int mBase = blockIdx.x * 128 + (wave >> 1) * 32;
    const int nWave = (wave & 1) * 64;
    const int n0 = blockIdx.y * 128;
    const unsigned bshBase = (unsigned)(uintptr_t)&Bsh[0];        // low 32b = LDS offset

    v8f acc[2][4] = {};
    staged_gemm_pass(rX, D0P, W0T, D0P, D0P, mBase, nWave, n0, lane, tid, Bsh, bshBase, acc); // r0 @ W0
    staged_gemm_pass(r2, H2, W1R, H2, H2, mBase, nWave, n0, lane, tid, Bsh, bshBase, acc);    // r2 @ W1^T

    const int mOff = (lane >= 16) ? 8 : 0;
#pragma unroll
    for (int i = 0; i < 2; ++i)
#pragma unroll
        for (int j = 0; j < 4; ++j)
#pragma unroll
            for (int v = 0; v < 8; ++v) {
                int m = mBase + i * 16 + mOff + v;
                int n = n0 + nWave + j * 16 + (lane & 15);
                float sOld = S[(size_t)m * SCOLS + n];
                float rp = (sOld >= 0.0f && sOld <= 1.0f) ? 1.0f : 0.0f;
                H[(size_t)m * H1 + n] = DECAY * sOld + DTC * rp * acc[i][j][v];
            }
}

// Layer-2 update (in place):  s2 = clip(0.8*s2 + 0.2*rho'(s2)*(r1@W1 + r3@W2^T))
__global__ void __launch_bounds__(256)
k_gemm_layer2(const _Float16* __restrict__ r1, const _Float16* __restrict__ W1T,
              const float* __restrict__ W2, float* __restrict__ S) {
    __shared__ _Float16 Bsh[2 * BSH_HALF];
    const int tid = threadIdx.x, lane = tid & 31, wave = tid >> 5;
    const int mBase = blockIdx.x * 128 + (wave >> 1) * 32;
    const int nWave = (wave & 1) * 64;
    const int n0 = blockIdx.y * 128;
    const unsigned bshBase = (unsigned)(uintptr_t)&Bsh[0];

    v8f acc[2][4] = {};
    staged_gemm_pass(r1, H1, W1T, H1, H1, mBase, nWave, n0, lane, tid, Bsh, bshBase, acc);    // r1 @ W1

    const int mOff = (lane >= 16) ? 8 : 0;
#pragma unroll
    for (int i = 0; i < 2; ++i)
#pragma unroll
        for (int j = 0; j < 4; ++j)
#pragma unroll
            for (int v = 0; v < 8; ++v) {
                int m = mBase + i * 16 + mOff + v;
                int n = n0 + nWave + j * 16 + (lane & 15);
                // small term: (r3 @ W2^T)[m][n] = sum_k clip(s3[m][k]) * W2[n][k]
                const float* s3 = S + (size_t)m * SCOLS + 2048;
                const float* w2r = W2 + (size_t)n * D3;
                float t = 0.0f;
#pragma unroll
                for (int k = 0; k < D3; ++k) t += clip01(s3[k]) * w2r[k];
                size_t idx = (size_t)m * SCOLS + 1024 + n;
                float sOld = S[idx];
                float rp = (sOld >= 0.0f && sOld <= 1.0f) ? 1.0f : 0.0f;
                S[idx] = clip01(DECAY * sOld + DTC * rp * (acc[i][j][v] + t));
            }
}

// Layer-3 update (tiny GEMM, in place):  s3 = clip(0.8*s3 + 0.2*rho'(s3)*(r2@W2))
__global__ void k_layer3(const _Float16* __restrict__ r2,
                         const float* __restrict__ W2, float* __restrict__ S) {
    int t = blockIdx.x * blockDim.x + threadIdx.x;        // B_ROWS * D3
    if (t >= B_ROWS * D3) return;
    int m = t / D3, n = t % D3;
    const _Float16* rr = r2 + (size_t)m * H2;
    float acc = 0.0f;
    for (int k = 0; k < H2; ++k) acc += (float)rr[k] * W2[(size_t)k * D3 + n];
    size_t idx = (size_t)m * SCOLS + 2048 + n;
    float sOld = S[idx];
    float rp = (sOld >= 0.0f && sOld <= 1.0f) ? 1.0f : 0.0f;
    S[idx] = clip01(DECAY * sOld + DTC * rp * acc);
}

// Column stats over H (8192 rows x 1024 cols), deterministic two-stage tree
__global__ void k_colstat_partial(const float* __restrict__ H,
                                  float* __restrict__ psum, float* __restrict__ psq) {
    int c = blockIdx.x * blockDim.x + threadIdx.x;        // gridDim.x = 4 -> 1024 cols
    int chunk = blockIdx.y;                               // 0..7, 1024 rows each
    float s = 0.0f, q = 0.0f;
    int r0 = chunk * 1024;
    for (int r = r0; r < r0 + 1024; ++r) {
        float v = H[(size_t)r * H1 + c];
        s += v; q += v * v;
    }
    psum[chunk * H1 + c] = s;
    psq [chunk * H1 + c] = q;
}

__global__ void k_colstat_final(const float* __restrict__ psum, const float* __restrict__ psq,
                                float* __restrict__ mean, float* __restrict__ inv) {
    int c = blockIdx.x * blockDim.x + threadIdx.x;        // 1024
    float s = 0.0f, q = 0.0f;
    for (int j = 0; j < 8; ++j) { s += psum[j * H1 + c]; q += psq[j * H1 + c]; }
    float mu = s * (1.0f / (float)B_ROWS);
    float var = fmaxf((q - s * mu) / (float)(B_ROWS - 1), 0.0f);  // ddof = 1
    mean[c] = mu;
    inv[c] = 1.0f / (sqrtf(var) + 1e-8f);
}

// s1 = clip((H - mean) / (std + 1e-8))
__global__ void k_norm(const float* __restrict__ H, const float* __restrict__ mean,
                       const float* __restrict__ inv, float* __restrict__ S) {
    int idx = blockIdx.x * blockDim.x + threadIdx.x;      // B_ROWS * H1
    int m = idx >> 10, c = idx & 1023;
    float v = (H[idx] - mean[c]) * inv[c];
    S[(size_t)m * SCOLS + c] = clip01(v);
}

// ---------------------------------------------------------------------------
extern "C" void kernel_launch(void* const* d_in, const int* in_sizes, int n_in,
                              void* d_out, int out_size, void* d_ws, size_t ws_size,
                              hipStream_t stream) {
    (void)in_sizes; (void)n_in; (void)out_size; (void)ws_size;
    const float* s_init = (const float*)d_in[0];
    const float* x      = (const float*)d_in[1];
    const float* W0     = (const float*)d_in[2];
    const float* W1     = (const float*)d_in[3];
    const float* W2     = (const float*)d_in[4];
    float* S = (float*)d_out;                             // state lives in d_out (in place)

    // Workspace carve-up (all offsets 256B aligned)
    char* w = (char*)d_ws;
    size_t off = 0;
    auto carve = [&](size_t bytes) { void* p = w + off; off += (bytes + 255) & ~(size_t)255; return p; };
    _Float16* rX  = (_Float16*)carve((size_t)B_ROWS * D0P * 2);   // 14.7 MB
    _Float16* r1  = (_Float16*)carve((size_t)B_ROWS * H1  * 2);   // 16.8 MB
    _Float16* r2  = (_Float16*)carve((size_t)B_ROWS * H2  * 2);   // 16.8 MB
    _Float16* r3p = (_Float16*)carve((size_t)B_ROWS * D3P * 2);   // 0.26 MB
    _Float16* W0T = (_Float16*)carve((size_t)H1 * D0P * 2);       // 1.8 MB
    _Float16* W1T = (_Float16*)carve((size_t)H1 * H2  * 2);       // 2 MB
    _Float16* W1R = (_Float16*)carve((size_t)H1 * H2  * 2);       // 2 MB
    float*    H   = (float*)carve((size_t)B_ROWS * H1 * 4);       // 33.6 MB
    float*    ps  = (float*)carve((size_t)8 * H1 * 4);
    float*    pq  = (float*)carve((size_t)8 * H1 * 4);
    float*    mu  = (float*)carve((size_t)H1 * 4);
    float*    iv  = (float*)carve((size_t)H1 * 4);

    // State init + one-time f16 prep
    hipMemcpyAsync(S, s_init, (size_t)B_ROWS * SCOLS * sizeof(float),
                   hipMemcpyDeviceToDevice, stream);
    k_prep_x <<<(B_ROWS * D0P + 255) / 256, 256, 0, stream>>>(x, rX);
    k_prep_w0<<<(H1 * D0P + 255) / 256, 256, 0, stream>>>(W0, W0T);
    k_prep_w1<<<(H1 * H2 + 255) / 256, 256, 0, stream>>>(W1, W1T, W1R);

    dim3 gemmGrid(B_ROWS / 128, H1 / 128);                // (64, 8)
    for (int it = 0; it < N_ITER; ++it) {
        k_actify<<<(B_ROWS * 2064 + 255) / 256, 256, 0, stream>>>(S, r1, r2, r3p);
        k_gemm_layer1<<<gemmGrid, 256, 0, stream>>>(rX, W0T, r2, W1R, S, H);
        k_gemm_layer2<<<gemmGrid, 256, 0, stream>>>(r1, W1T, W2, S);
        k_layer3<<<(B_ROWS * D3 + 255) / 256, 256, 0, stream>>>(r2, W2, S);
        k_colstat_partial<<<dim3(H1 / 256, 8), 256, 0, stream>>>(H, ps, pq);
        k_colstat_final<<<H1 / 256, 256, 0, stream>>>(ps, pq, mu, iv);
        k_norm<<<(B_ROWS * H1) / 256, 256, 0, stream>>>(H, mu, iv, S);
    }
}
